// Discriminator_23063974379767
// MI455X (gfx1250) — matrix operand order repaired and checked by reading, not compiled, not run on hardware
//
#include <hip/hip_runtime.h>
#include <hip/hip_bf16.h>

#define DEV __device__ __forceinline__

typedef __attribute__((ext_vector_type(16))) __bf16 v16bf;
typedef __attribute__((ext_vector_type(8)))  float  v8f;
typedef __attribute__((ext_vector_type(4)))  unsigned int uiv4;

// ---- problem sizes ----
constexpr int Nn   = 10000;
constexpr int HIN  = 128;
constexpr int Hh   = 512;
constexpr int TT   = 16;
constexpr int BB   = 512;
constexpr int MM   = BB * TT;      // 8192 rows of xc
constexpr int XLD  = 160;          // padded xc row (bf16): 128 proj + 1 weight + 31 zero
constexpr int KPAD = 160 + Hh;     // 672 padded K for the cell GEMM (21 tiles of 32)
constexpr int G4   = 4 * Hh;       // 2048 gate columns
constexpr int FRS  = 24;           // B-fragment lane stride (16 data + 8 pad bf16 = 48B)

// ---- async global->LDS helpers (ASYNCcnt path), guarded so compile never breaks ----
#if defined(__has_builtin)
#  if __has_builtin(__builtin_amdgcn_global_load_async_to_lds_b128) && \
      __has_builtin(__builtin_amdgcn_global_load_async_to_lds_b32)  && \
      __has_builtin(__builtin_amdgcn_s_wait_asynccnt)
#    define HAS_ASYNC_LDS 1
#  endif
#endif

typedef int v4i __attribute__((vector_size(16)));
typedef __attribute__((address_space(1))) int  g_i32;
typedef __attribute__((address_space(3))) int  l_i32;
typedef __attribute__((address_space(1))) v4i  g_v4i;
typedef __attribute__((address_space(3))) v4i  l_v4i;

DEV void cp_lds_b128(void* l, const void* g) {
#ifdef HAS_ASYNC_LDS
  __builtin_amdgcn_global_load_async_to_lds_b128(
      (g_v4i*)(unsigned long long)g,
      (l_v4i*)(unsigned int)(unsigned long long)l, 0, 0);
#else
  *(uiv4*)l = *(const uiv4*)g;
#endif
}
DEV void cp_lds_b32(void* l, const void* g) {
#ifdef HAS_ASYNC_LDS
  __builtin_amdgcn_global_load_async_to_lds_b32(
      (g_i32*)(unsigned long long)g,
      (l_i32*)(unsigned int)(unsigned long long)l, 0, 0);
#else
  *(unsigned int*)l = *(const unsigned int*)g;
#endif
}
DEV void cp_lds_wait() {
#ifdef HAS_ASYNC_LDS
  __builtin_amdgcn_s_wait_asynccnt(0);
#endif
}

DEV unsigned pack_bf16(float lo, float hi) {   // -> v_cvt_pk_bf16_f32
  union { __bf16 b[2]; unsigned u; } t;
  t.b[0] = (__bf16)lo; t.b[1] = (__bf16)hi;
  return t.u;
}

// B-fragment LDS position for element B(k, n) of a 32xN k-tile:
//   frag[nt = n>>4][lane = (n&15) + 16*(k>=16)][j = k&15]
DEV int bfrag_off_u32(int k2, int n) {  // k2 even; returns u32 index (j, j+1 pair)
  int ln = (n & 15) + ((k2 >= 16) ? 16 : 0);
  return (((n >> 4) * 32 + ln) * FRS + (k2 & 15)) >> 1;
}

// =====================================================================
// Kernel 1: xa = x[:, :, :N] @ W_down   (8192 x 10000) * (10000 x 128)
// 128x128 tile / workgroup, 8 waves. A: async fp32 -> LDS (x row stride is
// odd so only 4B-aligned -> b32 async). B: register-pipelined, converted to
// bf16 and stored in WMMA fragment order -> inner loop = 2 ds_b128 + wmma.
// =====================================================================
__global__ __launch_bounds__(256) void k_downproj(const float* __restrict__ x,
                                                  const float* __restrict__ Wd,
                                                  __bf16* __restrict__ xc) {
  __shared__ float As[2][128 * 32];                       // 16KB x2
  __shared__ alignas(16) __bf16 Bf[2][8 * 32 * FRS];      // 12KB x2
  const int tid = threadIdx.x, lane = tid & 31, wv = tid >> 5;
  const int m0 = blockIdx.x * 128;

  v8f acc[8];
#pragma unroll
  for (int i = 0; i < 8; ++i)
#pragma unroll
    for (int j = 0; j < 8; ++j) acc[i][j] = 0.0f;

  const int NKT = 313;  // 312 full K-tiles of 32 + one tail tile of 16
  float b0[2][4], b1[2][4];  // B pipeline registers (rows k2, k2+1; 4 cols)

  auto stageA = [&](int buf, int kt) {
    const int k0 = kt * 32;
    if (kt < 312) {
#pragma unroll
      for (int i = 0; i < 16; ++i) {
        int e = tid + i * 256;
        int r = e >> 5, cc = e & 31;
        cp_lds_b32(&As[buf][e], &x[(size_t)(m0 + r) * 10001 + k0 + cc]);
      }
    } else {  // tail: 16 valid K columns, zero-pad (synchronous)
      for (int i = tid; i < 128 * 32; i += 256) {
        int r = i >> 5, cc = i & 31;
        float v = 0.0f;
        if (cc < 16) v = x[(size_t)(m0 + r) * 10001 + k0 + cc];
        As[buf][i] = v;
      }
    }
  };
  auto loadB = [&](int kt) {  // global -> regs, one tile ahead
    const int k0 = kt * 32;
#pragma unroll
    for (int i = 0; i < 2; ++i) {
      int p = tid + i * 256;          // 512 work items: (16 even-k) x (32 n-quads)
      int n4 = (p & 31) * 4;
      int k2 = (p >> 5) * 2;
      bool ok = (k0 + k2 + 1) < Nn;   // guards the 16-row tail tile
#pragma unroll
      for (int q = 0; q < 4; ++q) {
        b0[i][q] = ok ? Wd[(size_t)(k0 + k2) * HIN + n4 + q] : 0.0f;
        b1[i][q] = ok ? Wd[(size_t)(k0 + k2 + 1) * HIN + n4 + q] : 0.0f;
      }
    }
  };
  auto commitB = [&](int buf) {  // regs -> bf16 fragment-layout LDS
#pragma unroll
    for (int i = 0; i < 2; ++i) {
      int p = tid + i * 256;
      int n4 = (p & 31) * 4;
      int k2 = (p >> 5) * 2;
#pragma unroll
      for (int q = 0; q < 4; ++q)
        ((unsigned*)Bf[buf])[bfrag_off_u32(k2, n4 + q)] = pack_bf16(b0[i][q], b1[i][q]);
    }
  };
  auto compute = [&](int buf) {
    const int koff = (lane >> 4) * 16;   // A 16x32 bf16 layout
    const float* Ar = &As[buf][(wv * 16 + (lane & 15)) * 32];
    v16bf a;
#pragma unroll
    for (int j = 0; j < 8; ++j) {
      a[j]     = (__bf16)Ar[koff + j];
      a[8 + j] = (__bf16)Ar[koff + 8 + j];
    }
#pragma unroll
    for (int nt = 0; nt < 8; ++nt) {
      union { uiv4 u[2]; v16bf v; } bb;
      const uiv4* Bp = (const uiv4*)&Bf[buf][(nt * 32 + lane) * FRS];
      bb.u[0] = Bp[0];
      bb.u[1] = Bp[1];
      acc[nt] = __builtin_amdgcn_wmma_f32_16x16x32_bf16(
          false, a, false, bb.v, (short)0, acc[nt], false, false);
    }
  };

  stageA(0, 0);
  loadB(0);
  for (int kt = 0; kt < NKT; ++kt) {
    commitB(kt & 1);
    cp_lds_wait();
    __syncthreads();
    if (kt + 1 < NKT) { stageA((kt + 1) & 1, kt + 1); loadB(kt + 1); }
    compute(kt & 1);
    __syncthreads();
  }

  // C/D layout: VGPR i -> M = wv*16 + i (+8 for lanes 16-31); lane&15 -> N
#pragma unroll
  for (int nt = 0; nt < 8; ++nt) {
    const int col = nt * 16 + (lane & 15);
#pragma unroll
    for (int i = 0; i < 8; ++i) {
      int mr = wv * 16 + i + ((lane >> 4) * 8);
      xc[(size_t)(m0 + mr) * XLD + col] = (__bf16)acc[nt][i];
    }
  }
  if (tid < 128) {  // weight column + zero padding
    size_t r = (size_t)(m0 + tid);
    xc[r * XLD + 128] = (__bf16)x[r * 10001 + 10000];
#pragma unroll
    for (int c = 129; c < 160; ++c) xc[r * XLD + c] = (__bf16)0.0f;
  }
}

// =====================================================================
// Kernel 2: one-time conversion W_cell (641 x 2048 fp32) -> padded bf16 (672 x 2048)
// =====================================================================
__global__ __launch_bounds__(256) void k_prep_wcell(const float* __restrict__ Wc,
                                                    __bf16* __restrict__ Wp) {
  int e = blockIdx.x * 256 + threadIdx.x;  // KPAD*G4 = 1376256
  if (e >= KPAD * G4) return;
  int r = e / G4, c = e % G4;
  float v = 0.0f;
  if (r < 129)       v = Wc[(size_t)r * G4 + c];
  else if (r >= 160) v = Wc[(size_t)(r - 31) * G4 + c];
  Wp[e] = (__bf16)v;
}

__global__ __launch_bounds__(256) void k_init_state(float* __restrict__ cst,
                                                    float* __restrict__ hf,
                                                    __bf16* __restrict__ hb) {
  int e = blockIdx.x * 256 + threadIdx.x;
  cst[e] = 0.0f; hf[e] = 0.0f; hb[e] = (__bf16)0.0f;
}

// =====================================================================
// Kernel 3: per-step gate GEMM: gates = [xc_t | h] @ W_pad + b_cell
// M=512, N=2048, K=672 (21 tiles of 32). A: async bf16->LDS (b128).
// B: register-pipelined into fragment-order LDS (no conversion needed).
// =====================================================================
__global__ __launch_bounds__(256) void k_gates(const __bf16* __restrict__ xc,
                                               const __bf16* __restrict__ hb,
                                               const __bf16* __restrict__ Wp,
                                               const float*  __restrict__ bcell,
                                               float* __restrict__ gates,
                                               int t) {
  __shared__ alignas(16) __bf16 As[2][128 * 32];      // 8KB x2
  __shared__ alignas(16) __bf16 Bf[2][8 * 32 * FRS];  // 12KB x2
  const int tid = threadIdx.x, lane = tid & 31, wv = tid >> 5;
  const int m0 = blockIdx.x * 128;   // batch rows
  const int n0 = blockIdx.y * 128;   // gate cols

  v8f acc[8];
#pragma unroll
  for (int i = 0; i < 8; ++i)
#pragma unroll
    for (int j = 0; j < 8; ++j) acc[i][j] = 0.0f;

  unsigned long long br0[2], br1[2];  // B pipeline registers (4 bf16 each)

  auto stageA = [&](int buf, int kt) {
    const int k0 = kt * 32;
    const __bf16* src; size_t stride, base;
    if (k0 < 160) { src = xc; base = (size_t)t * XLD + k0; stride = (size_t)TT * XLD; }
    else          { src = hb; base = (size_t)(k0 - 160);   stride = Hh; }
#pragma unroll
    for (int i = 0; i < 2; ++i) {
      int e = tid + i * 256;           // 512 b128 chunks (8 bf16), all 16B aligned
      int r = e >> 2, cc = (e & 3) * 8;
      cp_lds_b128(&As[buf][r * 32 + cc], &src[base + (size_t)(m0 + r) * stride + cc]);
    }
  };
  auto loadB = [&](int kt) {
    const int k0 = kt * 32;
#pragma unroll
    for (int i = 0; i < 2; ++i) {
      int p = tid + i * 256;
      int n4 = (p & 31) * 4;
      int k2 = (p >> 5) * 2;
      br0[i] = *(const unsigned long long*)&Wp[(size_t)(k0 + k2)     * G4 + n0 + n4];
      br1[i] = *(const unsigned long long*)&Wp[(size_t)(k0 + k2 + 1) * G4 + n0 + n4];
    }
  };
  auto commitB = [&](int buf) {
#pragma unroll
    for (int i = 0; i < 2; ++i) {
      int p = tid + i * 256;
      int n4 = (p & 31) * 4;
      int k2 = (p >> 5) * 2;
#pragma unroll
      for (int q = 0; q < 4; ++q) {
        unsigned lo = (unsigned)(br0[i] >> (16 * q)) & 0xffffu;
        unsigned hi = (unsigned)(br1[i] >> (16 * q)) & 0xffffu;
        ((unsigned*)Bf[buf])[bfrag_off_u32(k2, n4 + q)] = lo | (hi << 16);
      }
    }
  };
  auto compute = [&](int buf) {
    const int koff = (lane >> 4) * 16;
    const __bf16* Ar = &As[buf][(wv * 16 + (lane & 15)) * 32 + koff];
    union { uiv4 u[2]; v16bf v; } aa;
    aa.u[0] = *(const uiv4*)&Ar[0];    // K {koff..koff+7}
    aa.u[1] = *(const uiv4*)&Ar[8];    // K {koff+8..koff+15}
#pragma unroll
    for (int nt = 0; nt < 8; ++nt) {
      union { uiv4 u[2]; v16bf v; } bb;
      const uiv4* Bp = (const uiv4*)&Bf[buf][(nt * 32 + lane) * FRS];
      bb.u[0] = Bp[0];
      bb.u[1] = Bp[1];
      acc[nt] = __builtin_amdgcn_wmma_f32_16x16x32_bf16(
          false, aa.v, false, bb.v, (short)0, acc[nt], false, false);
    }
  };

  const int NKT = KPAD / 32;  // 21
  stageA(0, 0);
  loadB(0);
  for (int kt = 0; kt < NKT; ++kt) {
    commitB(kt & 1);
    cp_lds_wait();
    __syncthreads();
    if (kt + 1 < NKT) { stageA((kt + 1) & 1, kt + 1); loadB(kt + 1); }
    compute(kt & 1);
    __syncthreads();
  }

#pragma unroll
  for (int nt = 0; nt < 8; ++nt) {
    const int col = n0 + nt * 16 + (lane & 15);
    const float bc = bcell[col];
#pragma unroll
    for (int i = 0; i < 8; ++i) {
      int mr = wv * 16 + i + ((lane >> 4) * 8);
      gates[(size_t)(m0 + mr) * G4 + col] = acc[nt][i] + bc;
    }
  }
}

// =====================================================================
// Kernel 4: LSTM elementwise update (512 x 512)
// =====================================================================
__global__ __launch_bounds__(256) void k_lstm_elem(const float* __restrict__ gates,
                                                   float* __restrict__ cst,
                                                   float* __restrict__ hf,
                                                   __bf16* __restrict__ hb) {
  int e = blockIdx.x * 256 + threadIdx.x;  // 262144
  int row = e >> 9, col = e & 511;
  size_t g = (size_t)row * G4 + col;
  float ig = gates[g], cg = gates[g + 512], fg = gates[g + 1024], og = gates[g + 1536];
  float fs = 1.0f / (1.0f + __expf(-(fg + 1.0f)));
  float is = 1.0f / (1.0f + __expf(-ig));
  float os = 1.0f / (1.0f + __expf(-og));
  float cn = cst[e] * fs + is * tanhf(cg);
  float hn = os * tanhf(cn);
  cst[e] = cn; hf[e] = hn; hb[e] = (__bf16)hn;
}

// =====================================================================
// Kernel 5: pred = h @ W_out + b_out  (one wave per batch row)
// =====================================================================
__global__ __launch_bounds__(256) void k_head(const float* __restrict__ hf,
                                              const float* __restrict__ Wo,
                                              const float* __restrict__ bo,
                                              float* __restrict__ out) {
  const int lane = threadIdx.x & 31;
  const int row  = blockIdx.x * 8 + (threadIdx.x >> 5);
  float s = 0.0f;
  for (int j = lane; j < Hh; j += 32) s += hf[(size_t)row * Hh + j] * Wo[j];
#pragma unroll
  for (int m = 16; m >= 1; m >>= 1) s += __shfl_xor(s, m, 32);
  if (lane == 0) out[row] = s + bo[0];
}

// =====================================================================
extern "C" void kernel_launch(void* const* d_in, const int* in_sizes, int n_in,
                              void* d_out, int out_size, void* d_ws, size_t ws_size,
                              hipStream_t stream) {
  (void)in_sizes; (void)n_in; (void)out_size; (void)ws_size;
  const float* x  = (const float*)d_in[0];
  const float* Wd = (const float*)d_in[1];
  const float* Wc = (const float*)d_in[2];
  const float* bc = (const float*)d_in[3];
  const float* Wo = (const float*)d_in[4];
  const float* bo = (const float*)d_in[5];
  float* out = (float*)d_out;

  char* ws = (char*)d_ws;
  size_t off = 0;
  auto alloc = [&](size_t bytes) -> void* {
    void* p = ws + off;
    off = (off + bytes + 255) & ~(size_t)255;
    return p;
  };
  __bf16* xc    = (__bf16*)alloc((size_t)MM * XLD * 2);    // 2.62 MB
  __bf16* Wp    = (__bf16*)alloc((size_t)KPAD * G4 * 2);   // 2.75 MB
  __bf16* hb    = (__bf16*)alloc((size_t)BB * Hh * 2);
  float*  hf    = (float*) alloc((size_t)BB * Hh * 4);
  float*  cst   = (float*) alloc((size_t)BB * Hh * 4);
  float*  gates = (float*) alloc((size_t)BB * G4 * 4);     // 4 MB

  k_prep_wcell<<<(KPAD * G4 + 255) / 256, 256, 0, stream>>>(Wc, Wp);
  k_downproj<<<MM / 128, 256, 0, stream>>>(x, Wd, xc);
  k_init_state<<<(BB * Hh) / 256, 256, 0, stream>>>(cst, hf, hb);
  for (int t = 0; t < TT; ++t) {
    k_gates<<<dim3(BB / 128, G4 / 128), 256, 0, stream>>>(xc, hb, Wp, bc, gates, t);
    k_lstm_elem<<<(BB * Hh) / 256, 256, 0, stream>>>(gates, cst, hf, hb);
  }
  k_head<<<BB / 8, 256, 0, stream>>>(hf, Wo, bo, out);
}
